// BodyNet_52699248722028
// MI455X (gfx1250) — compile-verified, matrix operand order checked by reading, not run on hardware
//
#include <hip/hip_runtime.h>

typedef __attribute__((ext_vector_type(16))) _Float16 v16h;
typedef __attribute__((ext_vector_type(8)))  float    v8f;

#define BN_INV_C 0.9999950000374997f   // 1/sqrt(1+1e-5)

// ---------------------------------------------------------------------------
// Farthest point sampling: one block per batch, running min-dist kept in ws.
// ---------------------------------------------------------------------------
__global__ __launch_bounds__(256) void fps_kernel(
    const float* __restrict__ xyz, int N, int npoint,
    int* __restrict__ idx, float* __restrict__ distws)
{
    int b = blockIdx.x;
    const float* X = xyz + (size_t)b * N * 3;
    float* D = distws + (size_t)b * N;
    __shared__ float sval[256];
    __shared__ int   sidx[256];
    __shared__ int   sfar;
    for (int i = threadIdx.x; i < N; i += 256) D[i] = 1e10f;
    if (threadIdx.x == 0) sfar = 0;
    __syncthreads();
    for (int it = 0; it < npoint; ++it) {
        int far = sfar;
        if (threadIdx.x == 0) idx[b * npoint + it] = far;
        float cx = X[far * 3 + 0], cy = X[far * 3 + 1], cz = X[far * 3 + 2];
        float best = -1.0f; int bi = 0x7fffffff;
        for (int i = threadIdx.x; i < N; i += 256) {
            float dx = X[i * 3 + 0] - cx;
            float dy = X[i * 3 + 1] - cy;
            float dz = X[i * 3 + 2] - cz;
            float d = dx * dx + dy * dy + dz * dz;
            float dm = fminf(D[i], d);
            D[i] = dm;
            if (dm > best) { best = dm; bi = i; }
        }
        sval[threadIdx.x] = best; sidx[threadIdx.x] = bi;
        __syncthreads();
        for (int s = 128; s > 0; s >>= 1) {
            if (threadIdx.x < s) {
                float v2 = sval[threadIdx.x + s]; int i2 = sidx[threadIdx.x + s];
                if (v2 > sval[threadIdx.x] ||
                    (v2 == sval[threadIdx.x] && i2 < sidx[threadIdx.x])) {
                    sval[threadIdx.x] = v2; sidx[threadIdx.x] = i2;
                }
            }
            __syncthreads();
        }
        if (threadIdx.x == 0) sfar = sidx[0];
        __syncthreads();
    }
}

__global__ void gather_xyz_kernel(const float* __restrict__ xyz,
                                  const int* __restrict__ idx,
                                  float* __restrict__ out,
                                  int N, int S, int total)
{
    for (int t = blockIdx.x * blockDim.x + threadIdx.x; t < total;
         t += gridDim.x * blockDim.x) {
        int b = t / S;
        int gi = idx[t];
        out[t * 3 + 0] = xyz[((size_t)b * N + gi) * 3 + 0];
        out[t * 3 + 1] = xyz[((size_t)b * N + gi) * 3 + 1];
        out[t * 3 + 2] = xyz[((size_t)b * N + gi) * 3 + 2];
    }
}

// ---------------------------------------------------------------------------
// Ball query: serial index-ascending scan == sorted smallest indices.
// ---------------------------------------------------------------------------
__global__ void ball_query_kernel(const float* __restrict__ xyz,
                                  const float* __restrict__ new_xyz,
                                  int N, int S, int ns, float r2,
                                  int* __restrict__ gidx, int total)
{
    for (int t = blockIdx.x * blockDim.x + threadIdx.x; t < total;
         t += gridDim.x * blockDim.x) {
        int b = t / S;
        float cx = new_xyz[t * 3 + 0], cy = new_xyz[t * 3 + 1], cz = new_xyz[t * 3 + 2];
        const float* X = xyz + (size_t)b * N * 3;
        int* g = gidx + (size_t)t * ns;
        int cnt = 0;
        for (int k = 0; k < N && cnt < ns; ++k) {
            float dx = X[k * 3 + 0] - cx;
            float dy = X[k * 3 + 1] - cy;
            float dz = X[k * 3 + 2] - cz;
            if (dx * dx + dy * dy + dz * dz <= r2) g[cnt++] = k;
        }
        int first = (cnt > 0) ? g[0] : 0;
        for (int k = cnt; k < ns; ++k) g[k] = first;
        if (cnt == 0) g[0] = 0;
    }
}

// ---------------------------------------------------------------------------
// Build grouped feature matrix rows: [feat[gi] (Cf) | xyz[gi]-new_xyz (3) | 0 pad]
// ---------------------------------------------------------------------------
__global__ void group_points_kernel(const float* __restrict__ feat, int Cf,
                                    const float* __restrict__ xyz,
                                    const float* __restrict__ new_xyz,
                                    const int* __restrict__ gidx,
                                    _Float16* __restrict__ A,
                                    int N, int S, int ns, int K, int total)
{
    for (int r = blockIdx.x * blockDim.x + threadIdx.x; r < total;
         r += gridDim.x * blockDim.x) {
        int j = r % ns;
        int bs = r / ns;
        int b = bs / S;
        int gi = gidx[(size_t)bs * ns + j];
        const float* f = feat + ((size_t)b * N + gi) * Cf;
        const float* px = xyz + ((size_t)b * N + gi) * 3;
        const float* cx = new_xyz + (size_t)bs * 3;
        _Float16* row = A + (size_t)r * K;
        int c = 0;
        for (; c < Cf; ++c) row[c] = (_Float16)f[c];
        row[c++] = (_Float16)(px[0] - cx[0]);
        row[c++] = (_Float16)(px[1] - cx[1]);
        row[c++] = (_Float16)(px[2] - cx[2]);
        for (; c < K; ++c) row[c] = (_Float16)0.0f;
    }
}

// Concat two float feature blocks into padded half rows. b_ indexed row/bdiv
// (bdiv=S broadcasts a per-batch vector, bdiv=1 is row-aligned).
__global__ void concat_rows_kernel(const float* __restrict__ a, int Ca,
                                   const float* __restrict__ b_, int Cb, int bdiv,
                                   _Float16* __restrict__ A, int K, int rows)
{
    for (int r = blockIdx.x * blockDim.x + threadIdx.x; r < rows;
         r += gridDim.x * blockDim.x) {
        _Float16* row = A + (size_t)r * K;
        const float* pa = a + (size_t)r * Ca;
        const float* pb = b_ + (size_t)(r / bdiv) * Cb;
        int c = 0;
        for (int i = 0; i < Ca; ++i) row[c++] = (_Float16)pa[i];
        for (int i = 0; i < Cb; ++i) row[c++] = (_Float16)pb[i];
        for (; c < K; ++c) row[c] = (_Float16)0.0f;
    }
}

// ---------------------------------------------------------------------------
// Weight / vector packing to padded f16 / f32.
// ---------------------------------------------------------------------------
__global__ void pack_weights_kernel(const float* __restrict__ W, int Cout, int Cin,
                                    _Float16* __restrict__ Wh, int Kin, int total)
{
    for (int t = blockIdx.x * blockDim.x + threadIdx.x; t < total;
         t += gridDim.x * blockDim.x) {
        int o = t / Kin, k = t % Kin;
        Wh[t] = (o < Cout && k < Cin) ? (_Float16)W[(size_t)o * Cin + k] : (_Float16)0.0f;
    }
}

__global__ void pack_vec_kernel(const float* __restrict__ v, int C,
                                float* __restrict__ out, int Kout)
{
    for (int t = blockIdx.x * blockDim.x + threadIdx.x; t < Kout;
         t += gridDim.x * blockDim.x)
        out[t] = (t < C) ? v[t] : 0.0f;
}

// ---------------------------------------------------------------------------
// WMMA GEMM: C[M,Kout] = A[M,Kin] * W^T, fused bias + BN affine + ReLU.
//
// Macro-tile blocking: 128 threads = 4 waves per block. Block covers 64 rows
// (16 per wave) x up to 128 output channels (8 WMMA col-tiles). Per k-step:
//   - A tile (64x32) and W tile (<=128x32) staged in LDS by the whole block
//   - each wave builds its A fragment ONCE and reuses it for up to 8
//     v_wmma_f32_16x16x32_f16 into 8 v8f accumulators
// => ~43 flops/byte of global traffic vs ~1 for the naive 1-tile/wave form.
// Fragment layouts per CDNA5 ISA 7.12.2 (A: lane&15 = M row, lane>=16 holds
// upper K-halves; C: VGPR i -> M = i + 8*(lane>=16), N = lane&15).
// ---------------------------------------------------------------------------
__global__ __launch_bounds__(128) void gemm_wmma_kernel(
    const _Float16* __restrict__ A,   // M x Kin
    const _Float16* __restrict__ W,   // Kout x Kin (zero padded)
    const float* __restrict__ bias,
    const float* __restrict__ gamma,
    const float* __restrict__ beta,
    _Float16* __restrict__ outH,      // M x Kout
    float* __restrict__ outF,         // optional M x CoutF
    int M, int Kin, int Kout, int CoutF, int bnrelu)
{
    __shared__ __align__(16) _Float16 sA[64 * 32];    // 4 KB
    __shared__ __align__(16) _Float16 sW[128 * 32];   // 8 KB
    const int tid = threadIdx.x;
    const int lane = tid & 31;
    const int wv = tid >> 5;                 // wave id 0..3 -> row sub-tile
    const int rowBase = blockIdx.x * 64;
    const int colBase = blockIdx.y * 128;
    int rem = Kout - colBase;
    const int ncols = rem < 128 ? rem : 128; // multiple of 32
    const int ntiles = ncols >> 4;           // 2..8, uniform
    const int h = lane >> 4;                 // K-half selector
    const int m = lane & 15;                 // A row / W col within tile

    v8f acc[8];
    {
        v8f z = {};
#pragma unroll
        for (int i = 0; i < 8; ++i) acc[i] = z;
    }

    for (int k0 = 0; k0 < Kin; k0 += 32) {
        // ---- stage A tile: 64 rows x 32 halves, 2 x b128 per thread -------
        {
            int r = tid >> 1, c0 = (tid & 1) * 16;
            const uint4* gA = reinterpret_cast<const uint4*>(
                A + (size_t)(rowBase + r) * Kin + k0 + c0);
            uint4* shA = reinterpret_cast<uint4*>(&sA[r * 32 + c0]);
            shA[0] = gA[0]; shA[1] = gA[1];
            if (k0 + 32 < Kin) __builtin_prefetch(gA + 4, 0, 1);
        }
        // ---- stage W tile: ncols rows x 32 halves, 4 x b128 per thread ----
        if (tid < ncols) {
            const uint4* gW = reinterpret_cast<const uint4*>(
                W + (size_t)(colBase + tid) * Kin + k0);
            uint4* shW = reinterpret_cast<uint4*>(&sW[tid * 32]);
            shW[0] = gW[0]; shW[1] = gW[1]; shW[2] = gW[2]; shW[3] = gW[3];
            if (k0 + 32 < Kin) __builtin_prefetch(gW + 8, 0, 1);
        }
        __syncthreads();

        // ---- A fragment for this wave (reused across all col-tiles) -------
        union { v16h v; unsigned int u[8]; } ua;
#pragma unroll
        for (int vr = 0; vr < 8; ++vr) {
            int kb = ((vr & 4) ? 16 : 0) + h * 8 + (vr & 3) * 2;
            ua.u[vr] = *reinterpret_cast<const unsigned int*>(
                &sA[(wv * 16 + m) * 32 + kb]);
        }
        // ---- 8 guarded col-tiles, fully unrolled so acc stays in VGPRs ----
#pragma unroll
        for (int nt = 0; nt < 8; ++nt) {
            if (nt >= ntiles) break;   // uniform
            union { v16h v; unsigned int u[8]; } ub;
#pragma unroll
            for (int vr = 0; vr < 8; ++vr) {
                int kb = ((vr & 4) ? 16 : 0) + h * 8 + (vr & 3) * 2;
                ub.u[vr] = *reinterpret_cast<const unsigned int*>(
                    &sW[(nt * 16 + m) * 32 + kb]);
            }
            acc[nt] = __builtin_amdgcn_wmma_f32_16x16x32_f16(
                false, ua.v, false, ub.v, (short)0, acc[nt], false, false);
        }
        __syncthreads();
    }

    // ---- epilogue: bias + BN affine + ReLU, write f16 (+ optional f32) ----
#pragma unroll
    for (int nt = 0; nt < 8; ++nt) {
        if (nt >= ntiles) break;       // uniform
        const int cg = colBase + nt * 16 + m;
        const float bb = bias[cg];
        const float gg = gamma[cg];
        const float bt = beta[cg];
#pragma unroll
        for (int i = 0; i < 8; ++i) {
            int mr = rowBase + wv * 16 + i + 8 * h;
            float x = acc[nt][i] + bb;
            if (bnrelu) {
                x = gg * (x * BN_INV_C) + bt;
                x = x > 0.0f ? x : 0.0f;
            }
            outH[(size_t)mr * Kout + cg] = (_Float16)x;
            if (outF && cg < CoutF) outF[(size_t)mr * CoutF + cg] = x;
        }
    }
}

// ---------------------------------------------------------------------------
// Max over the nsample axis.
// ---------------------------------------------------------------------------
__global__ void rowmax_kernel(const _Float16* __restrict__ X,
                              int groups, int ns, int K, int Cvalid,
                              float* __restrict__ out, int ostride, int ooff)
{
    int total = groups * Cvalid;
    for (int t = blockIdx.x * blockDim.x + threadIdx.x; t < total;
         t += gridDim.x * blockDim.x) {
        int g = t / Cvalid, c = t % Cvalid;
        const _Float16* p = X + ((size_t)g * ns) * K + c;
        float mx = -1e30f;
        for (int j = 0; j < ns; ++j) {
            float v = (float)p[0];
            mx = v > mx ? v : mx;
            p += K;
        }
        out[(size_t)g * ostride + ooff + c] = mx;
    }
}

// ---------------------------------------------------------------------------
// Three-NN inverse-distance interpolation.
// ---------------------------------------------------------------------------
__global__ void three_interp_kernel(const float* __restrict__ xyz1,
                                    const float* __restrict__ xyz2,
                                    const float* __restrict__ feat2,
                                    float* __restrict__ out,
                                    int N1, int N2, int C, int total)
{
    for (int t = blockIdx.x * blockDim.x + threadIdx.x; t < total;
         t += gridDim.x * blockDim.x) {
        int b = t / N1;
        float px = xyz1[t * 3 + 0], py = xyz1[t * 3 + 1], pz = xyz1[t * 3 + 2];
        const float* X2 = xyz2 + (size_t)b * N2 * 3;
        float d0 = 1e30f, d1 = 1e30f, d2 = 1e30f;
        int i0 = 0, i1 = 0, i2 = 0;
        for (int j = 0; j < N2; ++j) {
            float dx = X2[j * 3 + 0] - px;
            float dy = X2[j * 3 + 1] - py;
            float dz = X2[j * 3 + 2] - pz;
            float d = dx * dx + dy * dy + dz * dz;
            if (d < d0)      { d2 = d1; i2 = i1; d1 = d0; i1 = i0; d0 = d; i0 = j; }
            else if (d < d1) { d2 = d1; i2 = i1; d1 = d; i1 = j; }
            else if (d < d2) { d2 = d; i2 = j; }
        }
        float w0 = 1.0f / (d0 + 1e-8f);
        float w1 = 1.0f / (d1 + 1e-8f);
        float w2 = 1.0f / (d2 + 1e-8f);
        float ws = w0 + w1 + w2;
        w0 /= ws; w1 /= ws; w2 /= ws;
        const float* f0 = feat2 + ((size_t)b * N2 + i0) * C;
        const float* f1 = feat2 + ((size_t)b * N2 + i1) * C;
        const float* f2 = feat2 + ((size_t)b * N2 + i2) * C;
        float* o = out + (size_t)t * C;
        for (int c = 0; c < C; ++c)
            o[c] = w0 * f0[c] + w1 * f1[c] + w2 * f2[c];
    }
}

// p1 = [one_hot(occ,2) | xyz | xyz]  (8 channels, float)
__global__ void build_p1_kernel(const int* __restrict__ occs,
                                const float* __restrict__ xyz,
                                float* __restrict__ p1, int total)
{
    for (int t = blockIdx.x * blockDim.x + threadIdx.x; t < total;
         t += gridDim.x * blockDim.x) {
        int occ = occs[t];
        float* row = p1 + (size_t)t * 8;
        row[0] = (occ == 0) ? 1.0f : 0.0f;
        row[1] = (occ == 1) ? 1.0f : 0.0f;
        float x = xyz[t * 3 + 0], y = xyz[t * 3 + 1], z = xyz[t * 3 + 2];
        row[2] = x; row[3] = y; row[4] = z;
        row[5] = x; row[6] = y; row[7] = z;
    }
}

// logits (B*N, 2) -> out (B, 2, N)
__global__ void transpose_out_kernel(const float* __restrict__ logits,
                                     float* __restrict__ out, int N, int total)
{
    for (int t = blockIdx.x * blockDim.x + threadIdx.x; t < total;
         t += gridDim.x * blockDim.x) {
        int n = t % N;
        int c = (t / N) & 1;
        int b = t / (2 * N);
        out[t] = logits[((size_t)b * N + n) * 2 + c];
    }
}

// ---------------------------------------------------------------------------
// Host orchestration
// ---------------------------------------------------------------------------
struct LayerP { const float* W; const float* b; const float* g; const float* be;
                int Cin; int Cout; };

static inline int ru32(int x) { return (x + 31) & ~31; }
static inline int cdiv(int a, int b) { return (a + b - 1) / b; }

static void run_chain(hipStream_t stream, const LayerP* L, int nL, int M,
                      _Float16*& cur, _Float16*& nxt,
                      _Float16* Wh, float* bscr, float* gscr, float* bescr,
                      float* outF, int CoutF)
{
    for (int i = 0; i < nL; ++i) {
        int Kin = ru32(L[i].Cin), Kout = ru32(L[i].Cout);
        int wt = Kin * Kout;
        pack_weights_kernel<<<cdiv(wt, 256), 256, 0, stream>>>(
            L[i].W, L[i].Cout, L[i].Cin, Wh, Kin, wt);
        pack_vec_kernel<<<cdiv(Kout, 256), 256, 0, stream>>>(L[i].b, L[i].Cout, bscr, Kout);
        pack_vec_kernel<<<cdiv(Kout, 256), 256, 0, stream>>>(
            L[i].g, L[i].g ? L[i].Cout : 0, gscr, Kout);
        pack_vec_kernel<<<cdiv(Kout, 256), 256, 0, stream>>>(
            L[i].be, L[i].be ? L[i].Cout : 0, bescr, Kout);
        float* of = (i == nL - 1) ? outF : nullptr;
        dim3 grid(M / 64, cdiv(Kout, 128));
        gemm_wmma_kernel<<<grid, 128, 0, stream>>>(
            cur, Wh, bscr, gscr, bescr, nxt, of, M, Kin, Kout, CoutF,
            L[i].g ? 1 : 0);
        _Float16* t = cur; cur = nxt; nxt = t;
    }
}

extern "C" void kernel_launch(void* const* d_in, const int* in_sizes, int n_in,
                              void* d_out, int out_size, void* d_ws, size_t ws_size,
                              hipStream_t stream)
{
    (void)in_sizes; (void)n_in; (void)out_size; (void)ws_size;
    const int B = 8, N = 4096;
    const float* xyz = (const float*)d_in[0];
    const int* occs = (const int*)d_in[1];

    // ---- parameter unpacking (dict insertion order; per layer W,b[,g,beta]) --
    int pi = 2;
    auto getL = [&](int cin, int cout, bool bn) {
        LayerP p;
        p.W = (const float*)d_in[pi++];
        p.b = (const float*)d_in[pi++];
        p.g = bn ? (const float*)d_in[pi++] : nullptr;
        p.be = bn ? (const float*)d_in[pi++] : nullptr;
        p.Cin = cin; p.Cout = cout;
        return p;
    };
    const int sa1_ch[3][4] = {{6,32,32,64},{6,64,64,128},{6,64,96,128}};
    const int sa2_ch[2][4] = {{323,128,128,256},{323,128,196,256}};
    const int sa3_ch[4] = {515,256,512,1024};
    LayerP sa1p[3][3], sa2p[2][3], sa3p[3], fp3p[2], fp2p[2], fp1p[2], headp[2];
    for (int br = 0; br < 3; ++br)
        for (int l = 0; l < 3; ++l)
            sa1p[br][l] = getL(sa1_ch[br][l], sa1_ch[br][l+1], true);
    for (int br = 0; br < 2; ++br)
        for (int l = 0; l < 3; ++l)
            sa2p[br][l] = getL(sa2_ch[br][l], sa2_ch[br][l+1], true);
    for (int l = 0; l < 3; ++l) sa3p[l] = getL(sa3_ch[l], sa3_ch[l+1], true);
    fp3p[0] = getL(1536, 256, true);  fp3p[1] = getL(256, 256, true);
    fp2p[0] = getL(576, 256, true);   fp2p[1] = getL(256, 128, true);
    fp1p[0] = getL(136, 128, true);   fp1p[1] = getL(128, 128, true);
    headp[0] = getL(128, 128, true);  headp[1] = getL(128, 2, false);

    // ---- workspace layout --------------------------------------------------
    size_t off = 0;
    char* ws = (char*)d_ws;
    auto alloc = [&](size_t bytes) -> void* {
        off = (off + 255) & ~(size_t)255;
        void* p = ws + off; off += bytes; return p;
    };
    const size_t CAP_HALF = (size_t)524288 * 128;   // largest M*Kpad (sa1 br2)
    _Float16* bufA = (_Float16*)alloc(CAP_HALF * 2);
    _Float16* bufB = (_Float16*)alloc(CAP_HALF * 2);
    _Float16* Wh   = (_Float16*)alloc((size_t)1 << 21);
    float* bscr  = (float*)alloc(1024 * 4);
    float* gscr  = (float*)alloc(1024 * 4);
    float* bescr = (float*)alloc(1024 * 4);
    int*   fpsidx = (int*)alloc(B * 512 * 4);
    float* distws = (float*)alloc((size_t)B * N * 4);
    float* l1_xyz = (float*)alloc((size_t)B * 512 * 3 * 4);
    float* l1_pts = (float*)alloc((size_t)B * 512 * 320 * 4);
    int*   gidx   = (int*)alloc((size_t)B * 512 * 128 * 4);
    float* l2_xyz = (float*)alloc((size_t)B * 128 * 3 * 4);
    float* l2_pts = (float*)alloc((size_t)B * 128 * 512 * 4);
    float* l3_pts = (float*)alloc((size_t)B * 1024 * 4);
    float* fp3_out = (float*)alloc((size_t)B * 128 * 256 * 4);
    float* interp2 = (float*)alloc((size_t)B * 512 * 256 * 4);
    float* fp2_out = (float*)alloc((size_t)B * 512 * 128 * 4);
    float* p1buf   = (float*)alloc((size_t)B * N * 8 * 4);
    float* interp1 = (float*)alloc((size_t)B * N * 128 * 4);
    float* logits  = (float*)alloc((size_t)B * N * 2 * 4);

    // ---- SA1: fps 4096 -> 512, three MSG branches --------------------------
    fps_kernel<<<B, 256, 0, stream>>>(xyz, N, 512, fpsidx, distws);
    gather_xyz_kernel<<<cdiv(B * 512, 256), 256, 0, stream>>>(
        xyz, fpsidx, l1_xyz, N, 512, B * 512);
    const float sa1_r[3] = {0.1f, 0.2f, 0.4f};
    const int sa1_ns[3] = {32, 64, 128};
    const int sa1_off[3] = {0, 64, 192};
    for (int br = 0; br < 3; ++br) {
        int ns = sa1_ns[br];
        int M = B * 512 * ns;
        ball_query_kernel<<<cdiv(B * 512, 256), 256, 0, stream>>>(
            xyz, l1_xyz, N, 512, ns, sa1_r[br] * sa1_r[br], gidx, B * 512);
        group_points_kernel<<<cdiv(M, 256), 256, 0, stream>>>(
            xyz, 3, xyz, l1_xyz, gidx, bufA, N, 512, ns, 32, M);
        _Float16* cur = bufA; _Float16* nxt = bufB;
        run_chain(stream, sa1p[br], 3, M, cur, nxt, Wh, bscr, gscr, bescr,
                  nullptr, 0);
        int Cl = sa1_ch[br][3];
        rowmax_kernel<<<cdiv(B * 512 * Cl, 256), 256, 0, stream>>>(
            cur, B * 512, ns, ru32(Cl), Cl, l1_pts, 320, sa1_off[br]);
    }

    // ---- SA2: fps 512 -> 128, two MSG branches -----------------------------
    fps_kernel<<<B, 256, 0, stream>>>(l1_xyz, 512, 128, fpsidx, distws);
    gather_xyz_kernel<<<cdiv(B * 128, 256), 256, 0, stream>>>(
        l1_xyz, fpsidx, l2_xyz, 512, 128, B * 128);
    const float sa2_r[2] = {0.4f, 0.8f};
    const int sa2_ns[2] = {64, 128};
    const int sa2_off[2] = {0, 256};
    for (int br = 0; br < 2; ++br) {
        int ns = sa2_ns[br];
        int M = B * 128 * ns;
        ball_query_kernel<<<cdiv(B * 128, 256), 256, 0, stream>>>(
            l1_xyz, l2_xyz, 512, 128, ns, sa2_r[br] * sa2_r[br], gidx, B * 128);
        group_points_kernel<<<cdiv(M, 256), 256, 0, stream>>>(
            l1_pts, 320, l1_xyz, l2_xyz, gidx, bufA, 512, 128, ns, ru32(323), M);
        _Float16* cur = bufA; _Float16* nxt = bufB;
        run_chain(stream, sa2p[br], 3, M, cur, nxt, Wh, bscr, gscr, bescr,
                  nullptr, 0);
        rowmax_kernel<<<cdiv(B * 128 * 256, 256), 256, 0, stream>>>(
            cur, B * 128, ns, 256, 256, l2_pts, 512, sa2_off[br]);
    }

    // ---- SA3: group_all ----------------------------------------------------
    {
        int M = B * 128;
        concat_rows_kernel<<<cdiv(M, 256), 256, 0, stream>>>(
            l2_xyz, 3, l2_pts, 512, 1, bufA, ru32(515), M);
        _Float16* cur = bufA; _Float16* nxt = bufB;
        run_chain(stream, sa3p, 3, M, cur, nxt, Wh, bscr, gscr, bescr,
                  nullptr, 0);
        rowmax_kernel<<<cdiv(B * 1024, 256), 256, 0, stream>>>(
            cur, B, 128, 1024, 1024, l3_pts, 1024, 0);
    }

    // ---- FP3: broadcast l3 onto l2 -----------------------------------------
    {
        int M = B * 128;
        concat_rows_kernel<<<cdiv(M, 256), 256, 0, stream>>>(
            l2_pts, 512, l3_pts, 1024, 128, bufA, 1536, M);
        _Float16* cur = bufA; _Float16* nxt = bufB;
        run_chain(stream, fp3p, 2, M, cur, nxt, Wh, bscr, gscr, bescr,
                  fp3_out, 256);
    }

    // ---- FP2: interp l2 -> l1 ----------------------------------------------
    {
        int M = B * 512;
        three_interp_kernel<<<cdiv(M, 256), 256, 0, stream>>>(
            l1_xyz, l2_xyz, fp3_out, interp2, 512, 128, 256, M);
        concat_rows_kernel<<<cdiv(M, 256), 256, 0, stream>>>(
            l1_pts, 320, interp2, 256, 1, bufA, 576, M);
        _Float16* cur = bufA; _Float16* nxt = bufB;
        run_chain(stream, fp2p, 2, M, cur, nxt, Wh, bscr, gscr, bescr,
                  fp2_out, 128);
    }

    // ---- FP1 + head: interp l1 -> l0, 136->128->128->128->2 ----------------
    {
        int M = B * N;
        build_p1_kernel<<<cdiv(M, 256), 256, 0, stream>>>(occs, xyz, p1buf, M);
        three_interp_kernel<<<cdiv(M, 256), 256, 0, stream>>>(
            xyz, l1_xyz, fp2_out, interp1, N, 512, 128, M);
        concat_rows_kernel<<<cdiv(M, 256), 256, 0, stream>>>(
            p1buf, 8, interp1, 128, 1, bufA, ru32(136), M);
        LayerP chain[4] = {fp1p[0], fp1p[1], headp[0], headp[1]};
        _Float16* cur = bufA; _Float16* nxt = bufB;
        run_chain(stream, chain, 4, M, cur, nxt, Wh, bscr, gscr, bescr,
                  logits, 2);
        transpose_out_kernel<<<cdiv(B * 2 * N, 256), 256, 0, stream>>>(
            logits, (float*)d_out, N, B * 2 * N);
    }
}
